// LSTM_38663295598727
// MI455X (gfx1250) — compile-verified, matrix operand order checked by reading, not run on hardware
//
#include <hip/hip_runtime.h>
#include <hip/hip_bf16.h>

// ---------------------------------------------------------------------------
// 2-layer LSTM (B=1024, T=2048, I=2, H=64) + FC(64->2), fused persistent
// kernel for gfx1250 (MI455X). wave32, WMMA bf16 16x16x32, weights in LDS.
// v2: fast rcp-based sigmoid/tanh (no IEEE div sequences), W_hh1 fragments
//     streamed from LDS to keep VGPR count < 256 (no s_set_vgpr_msb toggles).
// ---------------------------------------------------------------------------

typedef __attribute__((ext_vector_type(16))) __bf16 v16bf;
typedef __attribute__((ext_vector_type(8)))  __bf16 v8bf;
typedef __attribute__((ext_vector_type(8)))  float  v8f;

#define BATCH    1024
#define T_STEPS  2048
#define IN_DIM   2
#define HID      64
#define GATES    256      // 4*HID
#define TILE_B   16       // batch rows per block (one WMMA M tile)
#define NTHREADS 128      // 4 waves of 32

// Fast sigmoid: 1 v_exp + 1 add + 1 v_rcp.
__device__ __forceinline__ float sigmoidf_(float x) {
  return __builtin_amdgcn_rcpf(1.0f + __expf(-x));
}
// NaN-safe fast tanh: e = exp(-2|x|) <= 1, never overflows; rcp not div.
__device__ __forceinline__ float tanhf_(float x) {
  float e = __expf(-2.0f * fabsf(x));
  float r = (1.0f - e) * __builtin_amdgcn_rcpf(1.0f + e);
  return copysignf(r, x);
}

__device__ __forceinline__ v8f wmma_bf16(v16bf a, v16bf b, v8f c) {
  // D = A(16x32 bf16) x B(32x16 bf16) + C(16x16 f32)
  return __builtin_amdgcn_wmma_f32_16x16x32_bf16(
      /*neg_a=*/false, a, /*neg_b=*/false, b,
      /*c_mod=*/(short)0, c, /*reuse_a=*/false, /*reuse_b=*/false);
}

// Load one 16x32 bf16 WMMA fragment from LDS stored row-major [row][HID].
// ISA 16-bit A layout: lane L holds row (L%16); halves 0..7 = K kbase+8*(L/16)+0..7,
// halves 8..15 = K kbase+16+8*(L/16)+0..7  -> two 16B ds reads per lane.
// B layout mirrors this with "row" = N column index.
__device__ __forceinline__ v16bf load_frag(const __bf16* base, int row, int kbase, int half) {
  const __bf16* p = base + row * HID + kbase + 8 * half;
  v8bf lo = *(const v8bf*)(p);
  v8bf hi = *(const v8bf*)(p + 16);
  v16bf r;
#pragma unroll
  for (int i = 0; i < 8; ++i) { r[i] = lo[i]; r[i + 8] = hi[i]; }
  return r;
}

__global__ __launch_bounds__(NTHREADS)
void lstm2_wmma_kernel(const float* __restrict__ x,
                       const float* __restrict__ h0_in,
                       const float* __restrict__ c0_in,
                       const float* __restrict__ W_ih0,
                       const float* __restrict__ W_hh0,
                       const float* __restrict__ b_ih0,
                       const float* __restrict__ b_hh0,
                       const float* __restrict__ W_ih1,
                       const float* __restrict__ W_hh1,
                       const float* __restrict__ b_ih1,
                       const float* __restrict__ b_hh1,
                       const float* __restrict__ fc_w,
                       const float* __restrict__ fc_b,
                       float* __restrict__ out)
{
  extern __shared__ char smem[];
  __bf16* wt0  = (__bf16*)smem;            // [256][64]  W_hh0  (bf16)
  __bf16* wti1 = wt0  + GATES * HID;       // [256][64]  W_ih1
  __bf16* wt1  = wti1 + GATES * HID;       // [256][64]  W_hh1
  __bf16* h0b  = wt1  + GATES * HID;       // [2][16][64] double-buffered h (layer0)
  __bf16* h1b  = h0b  + 2 * TILE_B * HID;  // [2][16][64] double-buffered h (layer1)
  float*  xb   = (float*)(h1b + 2 * TILE_B * HID); // [16][2] x_t stage

  const int tid  = threadIdx.x;
  const int lane = tid & 31;
  const int wv   = tid >> 5;      // wave id 0..3
  const int half = lane >> 4;     // 0: lanes 0-15, 1: lanes 16-31
  const int l16  = lane & 15;
  const int bb0  = blockIdx.x * TILE_B;
  const int nrow = (wv << 4) + l16; // this lane's B-fragment row within a gate block

  // ---- stage weights into LDS (bf16, row-major [gate][k]) ----
  for (int i = tid; i < GATES * HID; i += NTHREADS) {
    wt0[i]  = (__bf16)W_hh0[i];
    wti1[i] = (__bf16)W_ih1[i];
    wt1[i]  = (__bf16)W_hh1[i];
  }
  // initial hidden states into parity-0 buffers
  for (int i = tid; i < TILE_B * HID; i += NTHREADS) {
    int m = i >> 6, j = i & 63;
    h0b[i] = (__bf16)h0_in[(bb0 + m) * HID + j];
    h1b[i] = (__bf16)h0_in[BATCH * HID + (bb0 + m) * HID + j];
  }

  // this lane's hidden column and its four gate rows
  const int ncol = nrow;
  const int gI = ncol, gF = 64 + ncol, gG = 128 + ncol, gO = 192 + ncol;
  // loop-invariant per-lane scalars (layer0 input proj has I=2 -> do it in VALU)
  const float wI0 = W_ih0[2*gI], wI1 = W_ih0[2*gI+1];
  const float wF0 = W_ih0[2*gF], wF1 = W_ih0[2*gF+1];
  const float wG0 = W_ih0[2*gG], wG1 = W_ih0[2*gG+1];
  const float wO0 = W_ih0[2*gO], wO1 = W_ih0[2*gO+1];
  const float bI0 = b_ih0[gI] + b_hh0[gI], bF0 = b_ih0[gF] + b_hh0[gF];
  const float bG0 = b_ih0[gG] + b_hh0[gG], bO0 = b_ih0[gO] + b_hh0[gO];
  const float bI1 = b_ih1[gI] + b_hh1[gI], bF1 = b_ih1[gF] + b_hh1[gF];
  const float bG1 = b_ih1[gG] + b_hh1[gG], bO1 = b_ih1[gO] + b_hh1[gO];

  // cell states resident in VGPRs (D-tile layout: M = r + 8*half, N = ncol)
  v8f c0v, c1v;
#pragma unroll
  for (int r = 0; r < 8; ++r) {
    int m = r + 8 * half;
    c0v[r] = c0_in[(bb0 + m) * HID + ncol];
    c1v[r] = c0_in[BATCH * HID + (bb0 + m) * HID + ncol];
  }

  __syncthreads();

  // hoist loop-invariant layer-0 recurrent B fragments into registers (64 VGPRs)
  v16bf B0[4][2];
#pragma unroll
  for (int g = 0; g < 4; ++g) {
#pragma unroll
    for (int k = 0; k < 2; ++k) {
      B0[g][k] = load_frag(wt0, g * 64 + nrow, 32 * k, half);
    }
  }

  int p = 0;
  for (int t = 0; t < T_STEPS; ++t) {
    // stage x[:, t, :] for this batch tile (wave 0) + prefetch ahead
    if (tid < TILE_B * IN_DIM) {
      int m = tid >> 1, d = tid & 1;
      xb[tid] = x[((bb0 + m) * T_STEPS + t) * IN_DIM + d];
    }
    if (tid < TILE_B && t + 8 < T_STEPS) {
      __builtin_prefetch(&x[((bb0 + tid) * T_STEPS + t + 8) * IN_DIM], 0, 1);
    }
    __syncthreads();

    // ================= layer 0: g = x_t@W_ih0^T + b0 + h0@W_hh0^T ==========
    const __bf16* h0cur = h0b + p * TILE_B * HID;
    v16bf A0 = load_frag(h0cur, l16, 0, half);
    v16bf A1 = load_frag(h0cur, l16, 32, half);

    v8f ai, af, ag, ao;
#pragma unroll
    for (int r = 0; r < 8; ++r) {
      int m = r + 8 * half;
      float x0 = xb[2 * m], x1 = xb[2 * m + 1];
      ai[r] = fmaf(x0, wI0, fmaf(x1, wI1, bI0));
      af[r] = fmaf(x0, wF0, fmaf(x1, wF1, bF0));
      ag[r] = fmaf(x0, wG0, fmaf(x1, wG1, bG0));
      ao[r] = fmaf(x0, wO0, fmaf(x1, wO1, bO0));
    }
    ai = wmma_bf16(A0, B0[0][0], ai); ai = wmma_bf16(A1, B0[0][1], ai);
    af = wmma_bf16(A0, B0[1][0], af); af = wmma_bf16(A1, B0[1][1], af);
    ag = wmma_bf16(A0, B0[2][0], ag); ag = wmma_bf16(A1, B0[2][1], ag);
    ao = wmma_bf16(A0, B0[3][0], ao); ao = wmma_bf16(A1, B0[3][1], ao);

    __bf16* h0nxt = h0b + (p ^ 1) * TILE_B * HID;
#pragma unroll
    for (int r = 0; r < 8; ++r) {
      float iv = sigmoidf_(ai[r]);
      float fv = sigmoidf_(af[r]);
      float gv = tanhf_(ag[r]);
      float ov = sigmoidf_(ao[r]);
      float c  = fmaf(fv, c0v[r], iv * gv);
      c0v[r] = c;
      float h = ov * tanhf_(c);
      int m = r + 8 * half;
      h0nxt[m * HID + ncol] = (__bf16)h;
    }
    __syncthreads();

    // ======== layer 1: g = h0new@W_ih1^T + b1 + h1@W_hh1^T =================
    v16bf Ax0 = load_frag(h0nxt, l16, 0, half);
    v16bf Ax1 = load_frag(h0nxt, l16, 32, half);
    const __bf16* h1cur = h1b + p * TILE_B * HID;
    v16bf Ah0 = load_frag(h1cur, l16, 0, half);
    v16bf Ah1 = load_frag(h1cur, l16, 32, half);

#pragma unroll
    for (int r = 0; r < 8; ++r) { ai[r] = bI1; af[r] = bF1; ag[r] = bG1; ao[r] = bO1; }

    // W_ih1 and W_hh1 fragments streamed from LDS each step (VGPR relief)
    {
      v16bf Wf;
      Wf = load_frag(wti1, 0 * 64 + nrow, 0,  half); ai = wmma_bf16(Ax0, Wf, ai);
      Wf = load_frag(wti1, 0 * 64 + nrow, 32, half); ai = wmma_bf16(Ax1, Wf, ai);
      Wf = load_frag(wti1, 1 * 64 + nrow, 0,  half); af = wmma_bf16(Ax0, Wf, af);
      Wf = load_frag(wti1, 1 * 64 + nrow, 32, half); af = wmma_bf16(Ax1, Wf, af);
      Wf = load_frag(wti1, 2 * 64 + nrow, 0,  half); ag = wmma_bf16(Ax0, Wf, ag);
      Wf = load_frag(wti1, 2 * 64 + nrow, 32, half); ag = wmma_bf16(Ax1, Wf, ag);
      Wf = load_frag(wti1, 3 * 64 + nrow, 0,  half); ao = wmma_bf16(Ax0, Wf, ao);
      Wf = load_frag(wti1, 3 * 64 + nrow, 32, half); ao = wmma_bf16(Ax1, Wf, ao);

      Wf = load_frag(wt1, 0 * 64 + nrow, 0,  half); ai = wmma_bf16(Ah0, Wf, ai);
      Wf = load_frag(wt1, 0 * 64 + nrow, 32, half); ai = wmma_bf16(Ah1, Wf, ai);
      Wf = load_frag(wt1, 1 * 64 + nrow, 0,  half); af = wmma_bf16(Ah0, Wf, af);
      Wf = load_frag(wt1, 1 * 64 + nrow, 32, half); af = wmma_bf16(Ah1, Wf, af);
      Wf = load_frag(wt1, 2 * 64 + nrow, 0,  half); ag = wmma_bf16(Ah0, Wf, ag);
      Wf = load_frag(wt1, 2 * 64 + nrow, 32, half); ag = wmma_bf16(Ah1, Wf, ag);
      Wf = load_frag(wt1, 3 * 64 + nrow, 0,  half); ao = wmma_bf16(Ah0, Wf, ao);
      Wf = load_frag(wt1, 3 * 64 + nrow, 32, half); ao = wmma_bf16(Ah1, Wf, ao);
    }

    __bf16* h1nxt = h1b + (p ^ 1) * TILE_B * HID;
#pragma unroll
    for (int r = 0; r < 8; ++r) {
      float iv = sigmoidf_(ai[r]);
      float fv = sigmoidf_(af[r]);
      float gv = tanhf_(ag[r]);
      float ov = sigmoidf_(ao[r]);
      float c  = fmaf(fv, c1v[r], iv * gv);
      c1v[r] = c;
      float h = ov * tanhf_(c);
      int m = r + 8 * half;
      h1nxt[m * HID + ncol] = (__bf16)h;
    }
    __syncthreads();
    p ^= 1;
  }

  // ---- final FC: out[b, c] = h1_last[b, :] . fc_w[c, :] + fc_b[c] ----
  const __bf16* hlast = h1b + p * TILE_B * HID;
  if (tid < TILE_B * 2) {
    int m = tid >> 1, cls = tid & 1;
    float s = fc_b[cls];
#pragma unroll
    for (int j = 0; j < HID; ++j)
      s = fmaf((float)hlast[m * HID + j], fc_w[cls * HID + j], s);
    out[(bb0 + m) * 2 + cls] = s;
  }
}

extern "C" void kernel_launch(void* const* d_in, const int* in_sizes, int n_in,
                              void* d_out, int out_size, void* d_ws, size_t ws_size,
                              hipStream_t stream) {
  (void)in_sizes; (void)n_in; (void)out_size; (void)d_ws; (void)ws_size;
  const float* x     = (const float*)d_in[0];
  const float* h0    = (const float*)d_in[1];
  const float* c0    = (const float*)d_in[2];
  const float* W_ih0 = (const float*)d_in[3];
  const float* W_hh0 = (const float*)d_in[4];
  const float* b_ih0 = (const float*)d_in[5];
  const float* b_hh0 = (const float*)d_in[6];
  const float* W_ih1 = (const float*)d_in[7];
  const float* W_hh1 = (const float*)d_in[8];
  const float* b_ih1 = (const float*)d_in[9];
  const float* b_hh1 = (const float*)d_in[10];
  const float* fc_w  = (const float*)d_in[11];
  const float* fc_b  = (const float*)d_in[12];
  float* out = (float*)d_out;

  size_t shmem = (size_t)(3 * GATES * HID + 4 * TILE_B * HID) * 2  // bf16 arrays
               + (size_t)(TILE_B * IN_DIM) * sizeof(float);        // x stage
  dim3 grid(BATCH / TILE_B), block(NTHREADS);
  lstm2_wmma_kernel<<<grid, block, shmem, stream>>>(
      x, h0, c0, W_ih0, W_hh0, b_ih0, b_hh0,
      W_ih1, W_hh1, b_ih1, b_hh1, fc_w, fc_b, out);
}
